// MultiHeadAttention_29283087024562
// MI455X (gfx1250) — matrix operand order repaired and checked by reading, not compiled, not run on hardware
//
#include <hip/hip_runtime.h>
#include <hip/hip_bf16.h>

// ---------- types ----------
typedef __attribute__((ext_vector_type(16))) __bf16 v16bf;
typedef __attribute__((ext_vector_type(8)))  __bf16 v8bf;
typedef __attribute__((ext_vector_type(8)))  float  v8f;

#define D_MODEL 1024
#define SEQ     2048
#define BATCH   2
#define HEADS   16
#define DK      64
#define M_TOTAL (BATCH*SEQ)   // 4096 rows for all GEMMs

// CDNA5 async global->LDS path (ASYNCcnt). The builtin's prototype (from the
// compiler diagnostic) is (int4 AS1* src_global, int4 AS3* dst_lds, imm offset, imm cpol).
#if defined(__AMDGCN__) && __has_builtin(__builtin_amdgcn_global_load_async_to_lds_b128) && __has_builtin(__builtin_amdgcn_s_wait_asynccnt)
#define ATTN_ASYNC 1
typedef int v4i_vs __attribute__((vector_size(4 * sizeof(int))));
typedef __attribute__((address_space(1))) v4i_vs* g_v4i_ptr;
typedef __attribute__((address_space(3))) v4i_vs* l_v4i_ptr;
#else
#define ATTN_ASYNC 0
#endif

// LDS row strides (padded to de-stride bank access; multiples of 16B)
#define KST_LD 72   // 32x64 tile rows padded to 72 elems (144 B)
#define VST_LD 40   // 64x32 tile rows padded to 40 elems (80 B)

// round-to-nearest-even f32 -> bf16
__device__ __forceinline__ __bf16 to_bf16(float f) {
  union { float f; unsigned u; } v; v.f = f;
  unsigned r = v.u + 0x7FFFu + ((v.u >> 16) & 1u);
  unsigned short h = (unsigned short)(r >> 16);
  __bf16 o; __builtin_memcpy(&o, &h, 2);
  return o;
}

// ---------- elementwise f32 -> bf16, 8 elems/thread ----------
__global__ void convert_f32_bf16(const float* __restrict__ in,
                                 __bf16* __restrict__ out, int n) {
  int i = (blockIdx.x * blockDim.x + threadIdx.x) * 8;
  if (i + 7 < n) {
    float4 a = *(const float4*)(in + i);
    float4 b = *(const float4*)(in + i + 4);
    v8bf o;
    o[0] = to_bf16(a.x); o[1] = to_bf16(a.y); o[2] = to_bf16(a.z); o[3] = to_bf16(a.w);
    o[4] = to_bf16(b.x); o[5] = to_bf16(b.y); o[6] = to_bf16(b.z); o[7] = to_bf16(b.w);
    *(v8bf*)(out + i) = o;
  }
}

// A-operand fragment (16x32 bf16, M x K): this code's k order (B loads use the same)
// lane L (row M = L&15), element i <-> K = (L>>4)*16 + i
__device__ __forceinline__ v16bf load_a_frag(const __bf16* __restrict__ base, int lane) {
  const int half = lane >> 4;
  v8bf lo = *(const v8bf*)(base + half * 16);
  v8bf hi = *(const v8bf*)(base + half * 16 + 8);
  v16bf a;
#pragma unroll
  for (int i = 0; i < 8; ++i) { a[i] = lo[i]; a[8 + i] = hi[i]; }
  return a;
}

// 16-elem contiguous fragment via two 16B loads (for 16B-aligned LDS rows)
__device__ __forceinline__ v16bf load_v16_16al(const __bf16* __restrict__ p) {
  v8bf lo = *(const v8bf*)p;
  v8bf hi = *(const v8bf*)(p + 8);
  v16bf r;
#pragma unroll
  for (int i = 0; i < 8; ++i) { r[i] = lo[i]; r[8 + i] = hi[i]; }
  return r;
}

// ---------- WMMA GEMM: C[m,n] = sum_k A[m,k]*W[n,k] + bias[n] ----------
// MODE 0: f32 Cout[m*1024+n]; MODE 1: bf16 [B,H,S,64]; MODE 2: bf16 [B,H,64,S]
template <int MODE>
__global__ __launch_bounds__(256)
void gemm_wmma(const __bf16* __restrict__ A,
               const __bf16* __restrict__ W,
               const float*  __restrict__ bias,
               void* __restrict__ Cout) {
  const int lane  = threadIdx.x & 31;
  const int wave  = threadIdx.x >> 5;
  const int waveM = wave >> 2;
  const int waveN = wave & 3;
  const int rowBase = blockIdx.x * 128 + waveM * 64;
  const int colBase = blockIdx.y * 128 + waveN * 32;
  const int half = lane >> 4;
  const int lrow = lane & 15;

  v8f acc[4][2];
#pragma unroll
  for (int i = 0; i < 4; ++i)
#pragma unroll
    for (int j = 0; j < 2; ++j)
#pragma unroll
      for (int r = 0; r < 8; ++r) acc[i][j][r] = 0.0f;

  for (int k0 = 0; k0 < D_MODEL; k0 += 32) {
    v16bf bfr[2];
#pragma unroll
    for (int j = 0; j < 2; ++j) {
      const __bf16* wp = W + (size_t)(colBase + j * 16 + lrow) * D_MODEL + k0 + half * 16;
      bfr[j] = *(const v16bf*)wp;
    }
#pragma unroll
    for (int i = 0; i < 4; ++i) {
      const __bf16* ap = A + (size_t)(rowBase + i * 16 + lrow) * D_MODEL + k0;
      if (k0 + 32 < D_MODEL) __builtin_prefetch(ap + 32, 0, 0);   // global_prefetch_b8
      v16bf afr = load_a_frag(ap, lane);
#pragma unroll
      for (int j = 0; j < 2; ++j) {
        acc[i][j] = __builtin_amdgcn_wmma_f32_16x16x32_bf16(
            false, afr, false, bfr[j], (short)0, acc[i][j], false, false);
      }
    }
  }

#pragma unroll
  for (int i = 0; i < 4; ++i) {
#pragma unroll
    for (int j = 0; j < 2; ++j) {
      const int gn = colBase + j * 16 + lrow;
      const float bv = bias[gn];
#pragma unroll
      for (int r = 0; r < 8; ++r) {
        const int gm = rowBase + i * 16 + half * 8 + r;
        const float val = acc[i][j][r] + bv;
        if (MODE == 0) {
          ((float*)Cout)[(size_t)gm * D_MODEL + gn] = val;
        } else {
          const int b = gm >> 11, s = gm & (SEQ - 1);
          const int h = gn >> 6,  d = gn & (DK - 1);
          __bf16* o = (__bf16*)Cout;
          if (MODE == 1)
            o[(((size_t)(b * HEADS + h)) * SEQ + s) * DK + d] = to_bf16(val);
          else
            o[(((size_t)(b * HEADS + h)) * DK + d) * SEQ + s] = to_bf16(val);
        }
      }
    }
  }
}

// ---------- flash attention: softmax(Q K^T / 8) V ----------
// Q,K: [B*H, S, 64] bf16.  Vt: [B*H, 64, S] bf16.  ctx: [B, S, 1024] bf16.
// Block = 128 threads (4 waves); wave owns 16 q-rows; 32-wide k blocks.
// K/V tiles shared across the block, double-buffered in LDS, filled by
// GLOBAL_LOAD_ASYNC_TO_LDS_B128 (ASYNCcnt) when the toolchain exposes it.
__global__ __launch_bounds__(128)
void flash_attn(const __bf16* __restrict__ Q,
                const __bf16* __restrict__ K,
                const __bf16* __restrict__ Vt,
                __bf16* __restrict__ ctx) {
#if ATTN_ASYNC
  __shared__ __align__(16) __bf16 Kst[2][32 * KST_LD];  // 32 rows x 64 d (padded)
  __shared__ __align__(16) __bf16 Vst[2][64 * VST_LD];  // 64 rows x 32 k (padded)
#endif
  __shared__ __align__(16) __bf16 Plds[4][16 * 40 + 8]; // per-wave 16x32 P tile

  const int lane = threadIdx.x & 31;
  const int wave = threadIdx.x >> 5;          // 0..3
  const int bh   = blockIdx.y;
  const int qbase = blockIdx.x * 64 + wave * 16;
  const int half = lane >> 4;
  const int lrow = lane & 15;

  const __bf16* Qh = Q  + (size_t)bh * SEQ * DK;
  const __bf16* Kh = K  + (size_t)bh * SEQ * DK;
  const __bf16* Vh = Vt + (size_t)bh * DK * SEQ;

  v16bf qa[2];
#pragma unroll
  for (int c = 0; c < 2; ++c)
    qa[c] = load_a_frag(Qh + (size_t)(qbase + lrow) * DK + c * 32, lane);

  float mrun[8], lrun[8];
  v8f o[4];
#pragma unroll
  for (int r = 0; r < 8; ++r) { mrun[r] = -1e30f; lrun[r] = 0.0f; }
#pragma unroll
  for (int t = 0; t < 4; ++t)
#pragma unroll
    for (int r = 0; r < 8; ++r) o[t][r] = 0.0f;

  __bf16* myP = &Plds[wave][0];
  const float scale = 0.125f;  // 1/sqrt(64)

#if ATTN_ASYNC
  // One block-tile (32 k-rows of K, 32 k-cols of V) = 16 wave-wide b128 async
  // transfers; each of the 4 waves issues 4.
  auto prefetch_tiles = [&](int buf, int kbase) {
#pragma unroll
    for (int j = 0; j < 2; ++j) {
      const int jj = wave * 2 + j;     // 0..7
      {
        const int row = jj * 4 + (lane >> 3);        // 0..31
        const int off = (lane & 7) * 8;              // 16B units
        const __bf16* gp = Kh + (size_t)(kbase + row) * DK + off;
        __bf16* lp = &Kst[buf][row * KST_LD + off];
        __builtin_amdgcn_global_load_async_to_lds_b128((g_v4i_ptr)gp, (l_v4i_ptr)lp, 0, 0);
      }
      {
        const int d   = jj * 8 + (lane >> 2);        // 0..63
        const int off = (lane & 3) * 8;
        const __bf16* gp = Vh + (size_t)d * SEQ + kbase + off;
        __bf16* lp = &Vst[buf][d * VST_LD + off];
        __builtin_amdgcn_global_load_async_to_lds_b128((g_v4i_ptr)gp, (l_v4i_ptr)lp, 0, 0);
      }
    }
  };
  prefetch_tiles(0, 0);
  __builtin_amdgcn_s_wait_asynccnt(0);
  __syncthreads();
#endif

  for (int kb = 0; kb < SEQ; kb += 32) {
    const int buf = (kb >> 5) & 1; (void)buf;
#if ATTN_ASYNC
    if (kb + 32 < SEQ) prefetch_tiles(buf ^ 1, kb + 32);   // overlap with compute
#endif

    // ---- scores: two 16x16 tiles over this 32-wide k block ----
    v8f s[2];
#pragma unroll
    for (int t = 0; t < 2; ++t) {
#pragma unroll
      for (int r = 0; r < 8; ++r) s[t][r] = 0.0f;
#pragma unroll
      for (int c = 0; c < 2; ++c) {
#if ATTN_ASYNC
        v16bf kfr = load_v16_16al(&Kst[buf][(t * 16 + lrow) * KST_LD + c * 32 + half * 16]);
#else
        v16bf kfr = *(const v16bf*)(Kh + (size_t)(kb + t * 16 + lrow) * DK + c * 32 + half * 16);
#endif
        s[t] = __builtin_amdgcn_wmma_f32_16x16x32_bf16(
            false, qa[c], false, kfr, (short)0, s[t], false, false);
      }
    }

    // ---- online softmax (row stats per accumulator slot r) ----
    float corr[8];
#pragma unroll
    for (int r = 0; r < 8; ++r) {
      const float a0 = s[0][r] * scale;
      const float a1 = s[1][r] * scale;
      float mx = fmaxf(a0, a1);
      mx = fmaxf(mx, __shfl_xor(mx, 1, 32));
      mx = fmaxf(mx, __shfl_xor(mx, 2, 32));
      mx = fmaxf(mx, __shfl_xor(mx, 4, 32));
      mx = fmaxf(mx, __shfl_xor(mx, 8, 32));
      const float mnew = fmaxf(mrun[r], mx);
      const float c  = __expf(mrun[r] - mnew);
      const float p0 = __expf(a0 - mnew);
      const float p1 = __expf(a1 - mnew);
      float rs = p0 + p1;
      rs += __shfl_xor(rs, 1, 32);
      rs += __shfl_xor(rs, 2, 32);
      rs += __shfl_xor(rs, 4, 32);
      rs += __shfl_xor(rs, 8, 32);
      lrun[r] = lrun[r] * c + rs;
      mrun[r] = mnew;
      corr[r] = c;
      myP[(half * 8 + r) * 40 + lrow]      = to_bf16(p0);
      myP[(half * 8 + r) * 40 + 16 + lrow] = to_bf16(p1);
    }
#pragma unroll
    for (int t = 0; t < 4; ++t)
#pragma unroll
      for (int r = 0; r < 8; ++r) o[t][r] *= corr[r];

    // ---- reload P as A-operand fragment (per-wave LDS, in-order DS) ----
    v16bf pa;
    {
      const __bf16* pr = myP + lrow * 40;
      v8bf lo = *(const v8bf*)(pr + half * 16);
      v8bf hi = *(const v8bf*)(pr + half * 16 + 8);
#pragma unroll
      for (int i = 0; i < 8; ++i) { pa[i] = lo[i]; pa[8 + i] = hi[i]; }
    }

    // ---- PV: o[t](16x16, d = t*16..) += P(16x32) x V(32x16) ----
#pragma unroll
    for (int t = 0; t < 4; ++t) {
#if ATTN_ASYNC
      v16bf vfr = load_v16_16al(&Vst[buf][(t * 16 + lrow) * VST_LD + half * 16]);
#else
      v16bf vfr = *(const v16bf*)(Vh + (size_t)(t * 16 + lrow) * SEQ + kb + half * 16);
#endif
      o[t] = __builtin_amdgcn_wmma_f32_16x16x32_bf16(
          false, pa, false, vfr, (short)0, o[t], false, false);
    }

#if ATTN_ASYNC
    if (kb + 32 < SEQ) {
      __builtin_amdgcn_s_wait_asynccnt(0);   // next buffer landed
      __syncthreads();                       // all waves done reading current buf
    }
#endif
  }

  // ---- normalize & store ctx (merge heads: [B,S,1024]) ----
  const int b = bh >> 4, h = bh & (HEADS - 1);
#pragma unroll
  for (int r = 0; r < 8; ++r) {
    const float inv = 1.0f / lrun[r];
    const int q = qbase + half * 8 + r;
#pragma unroll
    for (int t = 0; t < 4; ++t) {
      const int col = h * DK + t * 16 + lrow;
      ctx[((size_t)(b * SEQ + q)) * D_MODEL + col] = to_bf16(o[t][r] * inv);
    }
  }
}

// ---------- launcher ----------
extern "C" void kernel_launch(void* const* d_in, const int* in_sizes, int n_in,
                              void* d_out, int out_size, void* d_ws, size_t ws_size,
                              hipStream_t stream) {
  (void)in_sizes; (void)n_in; (void)out_size; (void)ws_size;
  const float* q  = (const float*)d_in[0];
  const float* k  = (const float*)d_in[1];
  const float* v  = (const float*)d_in[2];
  const float* Wq = (const float*)d_in[3];
  const float* bq = (const float*)d_in[4];
  const float* Wk = (const float*)d_in[5];
  const float* bk = (const float*)d_in[6];
  const float* Wv = (const float*)d_in[7];
  const float* bv = (const float*)d_in[8];
  const float* Wo = (const float*)d_in[9];
  const float* bo = (const float*)d_in[10];

  char* ws = (char*)d_ws;
  size_t off = 0;
  auto carve = [&](size_t bytes) {
    void* p = ws + off;
    off += (bytes + 255) & ~(size_t)255;
    return p;
  };
  const size_t ACT_N = (size_t)M_TOTAL * D_MODEL;   // 4M elems
  const size_t WGT_N = (size_t)D_MODEL * D_MODEL;   // 1M elems

  __bf16* qb  = (__bf16*)carve(ACT_N * 2);
  __bf16* kb  = (__bf16*)carve(ACT_N * 2);
  __bf16* vb  = (__bf16*)carve(ACT_N * 2);
  __bf16* Wqb = (__bf16*)carve(WGT_N * 2);
  __bf16* Wkb = (__bf16*)carve(WGT_N * 2);
  __bf16* Wvb = (__bf16*)carve(WGT_N * 2);
  __bf16* Wob = (__bf16*)carve(WGT_N * 2);
  __bf16* Qs  = (__bf16*)carve(ACT_N * 2);   // [B,H,S,64]
  __bf16* Ks  = (__bf16*)carve(ACT_N * 2);   // [B,H,S,64]
  __bf16* Vts = (__bf16*)carve(ACT_N * 2);   // [B,H,64,S]
  __bf16* ctx = (__bf16*)carve(ACT_N * 2);   // [B,S,1024]

  const int CT = 256;
  const int GA = (int)((ACT_N / 8 + CT - 1) / CT);
  const int GW = (int)((WGT_N / 8 + CT - 1) / CT);
  convert_f32_bf16<<<GA, CT, 0, stream>>>(q,  qb,  (int)ACT_N);
  convert_f32_bf16<<<GA, CT, 0, stream>>>(k,  kb,  (int)ACT_N);
  convert_f32_bf16<<<GA, CT, 0, stream>>>(v,  vb,  (int)ACT_N);
  convert_f32_bf16<<<GW, CT, 0, stream>>>(Wq, Wqb, (int)WGT_N);
  convert_f32_bf16<<<GW, CT, 0, stream>>>(Wk, Wkb, (int)WGT_N);
  convert_f32_bf16<<<GW, CT, 0, stream>>>(Wv, Wvb, (int)WGT_N);
  convert_f32_bf16<<<GW, CT, 0, stream>>>(Wo, Wob, (int)WGT_N);

  dim3 ggrid(M_TOTAL / 128, D_MODEL / 128);   // 32 x 8
  gemm_wmma<1><<<ggrid, 256, 0, stream>>>(qb,  Wqb, bq, (void*)Qs);
  gemm_wmma<1><<<ggrid, 256, 0, stream>>>(kb,  Wkb, bk, (void*)Ks);
  gemm_wmma<2><<<ggrid, 256, 0, stream>>>(vb,  Wvb, bv, (void*)Vts);

  dim3 agrid(SEQ / 64, BATCH * HEADS);        // 32 x 32
  flash_attn<<<agrid, 128, 0, stream>>>(Qs, Ks, Vts, ctx);

  gemm_wmma<0><<<ggrid, 256, 0, stream>>>(ctx, Wob, bo, d_out);
}
